// HomogeneousGCN_8727373545900
// MI455X (gfx1250) — matrix operand order repaired and checked by reading, not compiled, not run on hardware
//
#include <hip/hip_runtime.h>

#define HD 32    // hidden width
#define FF 128   // input feature width
#define GG 512   // number of graphs
#define OO 16    // output width

typedef __attribute__((ext_vector_type(16))) __bf16 v16bf;
typedef __attribute__((ext_vector_type(8)))  float  v8f;

// C[M x NCOLS] = A[M x KDIM] * W[KDIM x NCOLS] (+ bias), bf16 WMMA, f32 accum.
// One wave per 16-row tile. M must be a multiple of 16.
template <int KDIM, int NCOLS>
__global__ void wmma_gemm(const float* __restrict__ A, const float* __restrict__ W,
                          const float* __restrict__ bias, float* __restrict__ C, int M) {
  constexpr int NT = NCOLS / 16;
  constexpr int KSTEPS = KDIM / 32;
  const int lane  = threadIdx.x & 31;
  const int wave  = threadIdx.x >> 5;
  const int mtile = blockIdx.x * (blockDim.x >> 5) + wave;
  if (mtile * 16 >= M) return;
  const int half = lane >> 4;                 // which half-wave
  const int mrow = mtile * 16 + (lane & 15);  // A row for this lane
  const int ncol = lane & 15;                 // B/C column for this lane

  v8f zero = {};
  v8f acc[NT];
#pragma unroll
  for (int i = 0; i < NT; ++i) acc[i] = zero;

#pragma unroll
  for (int kt = 0; kt < KSTEPS; ++kt) {
    if (kt + 1 < KSTEPS)
      __builtin_prefetch(A + (size_t)mrow * KDIM + (size_t)(kt + 1) * 32, 0, 0);
    // A fragment: 16-bit A 16x32 layout (ISA 7.12.2):
    //   lanes 0-15 (M=lane): dwords 0-3 hold K=0..7, dwords 4-7 hold K=16..23
    //   lanes 16-31 (M=lane-16): dwords 0-3 hold K=8..15, dwords 4-7 hold K=24..31
    v16bf a;
#pragma unroll
    for (int j = 0; j < 8; ++j) {
      int kb = kt * 32 + ((j < 4) ? 0 : 16) + half * 8 + (j & 3) * 2;
      a[2 * j]     = (__bf16)A[(size_t)mrow * KDIM + kb];
      a[2 * j + 1] = (__bf16)A[(size_t)mrow * KDIM + kb + 1];
    }
#pragma unroll
    for (int nt = 0; nt < NT; ++nt) {
      // B fragment: 16-bit B 32x16 layout:
      //   lanes 0-15 (N=lane): dwords 0-3 hold K=0..7, dwords 4-7 hold K=16..23; lanes 16-31 (N=lane-16): K=8..15, K=24..31
      v16bf b;
#pragma unroll
      for (int j = 0; j < 8; ++j) {
        int kb = kt * 32 + ((j < 4) ? 0 : 16) + half * 8 + (j & 3) * 2;
        b[2 * j]     = (__bf16)W[(size_t)kb * NCOLS + nt * 16 + ncol];
        b[2 * j + 1] = (__bf16)W[(size_t)(kb + 1) * NCOLS + nt * 16 + ncol];
      }
      acc[nt] = __builtin_amdgcn_wmma_f32_16x16x32_bf16(
          false, a, false, b, (short)0, acc[nt], false, false);
    }
  }

  // C/D layout: VGPR r, lanes 0-15 -> row r, lanes 16-31 -> row r+8
#pragma unroll
  for (int nt = 0; nt < NT; ++nt) {
#pragma unroll
    for (int r = 0; r < 8; ++r) {
      int row = mtile * 16 + half * 8 + r;
      int col = nt * 16 + ncol;
      float v = acc[nt][r];
      if (bias) v += bias[col];
      C[(size_t)row * NCOLS + col] = v;
    }
  }
}

__global__ void fill_f32(float* __restrict__ p, float v, long long n) {
  long long i = (long long)blockIdx.x * blockDim.x + threadIdx.x;
  if (i < n) p[i] = v;
}

__global__ void deg_accum(const int* __restrict__ dst, float* __restrict__ deg, long long e) {
  long long i = (long long)blockIdx.x * blockDim.x + threadIdx.x;
  if (i < e) atomicAdd(&deg[dst[i]], 1.0f);
}

__global__ void deg_to_dinv(float* __restrict__ deg, long long n) {
  long long i = (long long)blockIdx.x * blockDim.x + threadIdx.x;
  if (i < n) deg[i] = rsqrtf(deg[i]);  // deg >= 1 due to self-loops
}

// out[i,f] = h[i,f]*dinv[i]^2 (+ bias[f]) : self-loop term, also initializes out
__global__ void agg_self(const float* __restrict__ h, const float* __restrict__ dinv,
                         const float* __restrict__ bias, float* __restrict__ out, long long n) {
  long long t = (long long)blockIdx.x * blockDim.x + threadIdx.x;
  long long i = t >> 5;
  int f = (int)(t & 31);
  if (i >= n) return;
  float di = dinv[i];
  float v = h[i * HD + f] * di * di;
  if (bias) v += bias[f];
  out[i * HD + f] = v;
}

// out[dst] += h[src] * dinv[src]*dinv[dst] ; one wave of lanes = one edge row
__global__ void agg_edges(const float* __restrict__ h, const int* __restrict__ src,
                          const int* __restrict__ dst, const float* __restrict__ dinv,
                          float* __restrict__ out, long long e) {
  long long t = (long long)blockIdx.x * blockDim.x + threadIdx.x;
  long long idx = t >> 5;
  int f = (int)(t & 31);
  if (idx >= e) return;
  int s = src[idx], d = dst[idx];
  float w = dinv[s] * dinv[d];
  atomicAdd(&out[(size_t)d * HD + f], h[(size_t)s * HD + f] * w);
}

// per-column sum and sum of squares over n rows -> sums[0..31], sums[32..63]
__global__ void bn_stats(const float* __restrict__ x, float* __restrict__ sums, int n) {
  __shared__ float s0[256];
  __shared__ float s1[256];
  int f = threadIdx.x & 31;
  int rg = threadIdx.x >> 5;  // 0..7 row groups per block
  float a = 0.f, b = 0.f;
  for (int i = blockIdx.x * 8 + rg; i < n; i += gridDim.x * 8) {
    float v = x[(size_t)i * HD + f];
    a += v;
    b += v * v;
  }
  s0[threadIdx.x] = a;
  s1[threadIdx.x] = b;
  __syncthreads();
  if (threadIdx.x < 32) {
    float ta = 0.f, tb = 0.f;
#pragma unroll
    for (int r = 0; r < 8; ++r) { ta += s0[r * 32 + f]; tb += s1[r * 32 + f]; }
    atomicAdd(&sums[f], ta);
    atomicAdd(&sums[32 + f], tb);
  }
}

// scale/shift so that y = x*ss[f] + ss[32+f] equals (x-m)*rsqrt(v+eps)*g + be
__global__ void bn_finalize(const float* __restrict__ sums, const float* __restrict__ g,
                            const float* __restrict__ be, float* __restrict__ ss, float n) {
  int f = threadIdx.x;  // 32 threads
  float m  = sums[f] / n;
  float v  = sums[32 + f] / n - m * m;
  float iv = rsqrtf(v + 1e-5f);
  float sc = iv * g[f];
  ss[f] = sc;
  ss[32 + f] = be[f] - m * sc;
}

__global__ void bn_apply_relu(const float* __restrict__ x, const float* __restrict__ ss,
                              float* __restrict__ y, long long count) {
  long long i = (long long)blockIdx.x * blockDim.x + threadIdx.x;
  if (i >= count) return;
  int f = (int)(i & 31);
  y[i] = fmaxf(x[i] * ss[f] + ss[32 + f], 0.0f);
}

__global__ void pool_count(const int* __restrict__ batch, float* __restrict__ cnt, long long n) {
  long long i = (long long)blockIdx.x * blockDim.x + threadIdx.x;
  if (i < n) atomicAdd(&cnt[batch[i]], 1.0f);
}

__global__ void pool_sum(const float* __restrict__ z, const int* __restrict__ batch,
                         float* __restrict__ pooled, long long n) {
  long long t = (long long)blockIdx.x * blockDim.x + threadIdx.x;
  long long i = t >> 5;
  int f = (int)(t & 31);
  if (i >= n) return;
  atomicAdd(&pooled[(size_t)batch[i] * HD + f], z[i * HD + f]);
}

__global__ void pool_div(float* __restrict__ pooled, const float* __restrict__ cnt, long long count) {
  long long t = (long long)blockIdx.x * blockDim.x + threadIdx.x;
  if (t >= count) return;
  pooled[t] /= fmaxf(cnt[t >> 5], 1.0f);
}

static inline unsigned nblk(long long total, int per) {
  return (unsigned)((total + per - 1) / per);
}

extern "C" void kernel_launch(void* const* d_in, const int* in_sizes, int n_in,
                              void* d_out, int out_size, void* d_ws, size_t ws_size,
                              hipStream_t stream) {
  const float* x    = (const float*)d_in[0];
  const int*   ei   = (const int*)d_in[1];
  const int*   batch= (const int*)d_in[2];
  const float* W1   = (const float*)d_in[3];
  const float* g1   = (const float*)d_in[5];
  const float* be1  = (const float*)d_in[6];
  const float* W2   = (const float*)d_in[7];
  const float* g2   = (const float*)d_in[9];
  const float* be2  = (const float*)d_in[10];
  const float* W3   = (const float*)d_in[11];
  const float* b3   = (const float*)d_in[12];
  const float* mW1  = (const float*)d_in[13];
  const float* mb1  = (const float*)d_in[14];
  const float* mg1  = (const float*)d_in[15];
  const float* mbe1 = (const float*)d_in[16];
  const float* mW2  = (const float*)d_in[17];
  const float* mb2  = (const float*)d_in[18];
  const float* mg2  = (const float*)d_in[19];
  const float* mbe2 = (const float*)d_in[20];
  const float* mW3  = (const float*)d_in[21];
  const float* mb3  = (const float*)d_in[22];

  const long long N = in_sizes[0] / FF;
  const long long E = in_sizes[1] / 2;
  const int* srcp = ei;
  const int* dstp = ei + E;

  float* ws     = (float*)d_ws;
  float* dinv   = ws;  ws += N;
  float* bufA   = ws;  ws += N * HD;
  float* bufB   = ws;  ws += N * HD;
  float* stats  = ws;  ws += 64;
  float* ss     = ws;  ws += 64;
  float* pooled = ws;  ws += (long long)GG * HD;
  float* cnt    = ws;  ws += GG;
  float* m1     = ws;  ws += (long long)GG * HD;
  float* m2     = ws;  ws += (long long)GG * HD;

  // ---- GCN normalization: deg (with self loop) -> dinv ----
  fill_f32<<<nblk(N, 256), 256, 0, stream>>>(dinv, 1.0f, N);
  deg_accum<<<nblk(E, 256), 256, 0, stream>>>(dstp, dinv, E);
  deg_to_dinv<<<nblk(N, 256), 256, 0, stream>>>(dinv, N);

  // ---- Layer 1: h = X @ W1 ; aggregate ; BN ; ReLU ----
  wmma_gemm<FF, HD><<<nblk(N / 16, 4), 128, 0, stream>>>(x, W1, nullptr, bufA, (int)N);
  agg_self<<<nblk(N * HD, 256), 256, 0, stream>>>(bufA, dinv, nullptr, bufB, N);
  agg_edges<<<nblk(E * HD, 256), 256, 0, stream>>>(bufA, srcp, dstp, dinv, bufB, E);
  fill_f32<<<1, 64, 0, stream>>>(stats, 0.0f, 64);
  bn_stats<<<512, 256, 0, stream>>>(bufB, stats, (int)N);
  bn_finalize<<<1, 32, 0, stream>>>(stats, g1, be1, ss, (float)N);
  bn_apply_relu<<<nblk(N * HD, 256), 256, 0, stream>>>(bufB, ss, bufA, N * HD);

  // ---- Layer 2 ----
  wmma_gemm<HD, HD><<<nblk(N / 16, 4), 128, 0, stream>>>(bufA, W2, nullptr, bufB, (int)N);
  agg_self<<<nblk(N * HD, 256), 256, 0, stream>>>(bufB, dinv, nullptr, bufA, N);
  agg_edges<<<nblk(E * HD, 256), 256, 0, stream>>>(bufB, srcp, dstp, dinv, bufA, E);
  fill_f32<<<1, 64, 0, stream>>>(stats, 0.0f, 64);
  bn_stats<<<512, 256, 0, stream>>>(bufA, stats, (int)N);
  bn_finalize<<<1, 32, 0, stream>>>(stats, g2, be2, ss, (float)N);
  bn_apply_relu<<<nblk(N * HD, 256), 256, 0, stream>>>(bufA, ss, bufB, N * HD);

  // ---- Layer 3 (no BN/act), bias b3 applied in self-loop init ----
  wmma_gemm<HD, HD><<<nblk(N / 16, 4), 128, 0, stream>>>(bufB, W3, nullptr, bufA, (int)N);
  agg_self<<<nblk(N * HD, 256), 256, 0, stream>>>(bufA, dinv, b3, bufB, N);
  agg_edges<<<nblk(E * HD, 256), 256, 0, stream>>>(bufA, srcp, dstp, dinv, bufB, E);

  // ---- scatter-mean pooling ----
  fill_f32<<<nblk((long long)GG * HD + GG, 256), 256, 0, stream>>>(pooled, 0.0f,
                                                                   (long long)GG * HD + GG);
  pool_sum<<<nblk(N * HD, 256), 256, 0, stream>>>(bufB, batch, pooled, N);
  pool_count<<<nblk(N, 256), 256, 0, stream>>>(batch, cnt, N);
  pool_div<<<nblk((long long)GG * HD, 256), 256, 0, stream>>>(pooled, cnt, (long long)GG * HD);

  // ---- MLP head on [G=512, 32] ----
  wmma_gemm<HD, HD><<<nblk(GG / 16, 4), 128, 0, stream>>>(pooled, mW1, mb1, m1, GG);
  fill_f32<<<1, 64, 0, stream>>>(stats, 0.0f, 64);
  bn_stats<<<64, 256, 0, stream>>>(m1, stats, GG);
  bn_finalize<<<1, 32, 0, stream>>>(stats, mg1, mbe1, ss, (float)GG);
  bn_apply_relu<<<nblk((long long)GG * HD, 256), 256, 0, stream>>>(m1, ss, m1, (long long)GG * HD);

  wmma_gemm<HD, HD><<<nblk(GG / 16, 4), 128, 0, stream>>>(m1, mW2, mb2, m2, GG);
  fill_f32<<<1, 64, 0, stream>>>(stats, 0.0f, 64);
  bn_stats<<<64, 256, 0, stream>>>(m2, stats, GG);
  bn_finalize<<<1, 32, 0, stream>>>(stats, mg2, mbe2, ss, (float)GG);
  bn_apply_relu<<<nblk((long long)GG * HD, 256), 256, 0, stream>>>(m2, ss, m2, (long long)GG * HD);

  wmma_gemm<HD, OO><<<nblk(GG / 16, 4), 128, 0, stream>>>(m2, mW3, mb3, (float*)d_out, GG);
}
